// LIIF_37572373906043
// MI455X (gfx1250) — matrix operand order repaired and checked by reading, not compiled, not run on hardware
//
#include <hip/hip_runtime.h>
#include <hip/hip_bf16.h>

typedef __attribute__((ext_vector_type(16))) __bf16 v16bf;
typedef __attribute__((ext_vector_type(8)))  __bf16 v8bf;
typedef __attribute__((ext_vector_type(8)))  float  v8f;
typedef __attribute__((ext_vector_type(4)))  float  v4f;
typedef __attribute__((ext_vector_type(4)))  int    v4i;

#define HW 96
#define PW 98            // padded width/height
#define CH 64
#define NPIX (HW*HW)     // 9216
#define PADPIX (PW*PW)   // 9604
#define KCONV 576        // 64*9
#define Q 65536

#if defined(__HIP_DEVICE_COMPILE__) && __has_builtin(__builtin_amdgcn_global_load_async_to_lds_b128)
#define HAVE_ASYNC_LDS 1
#else
#define HAVE_ASYNC_LDS 0
#endif

// ---------------- WMMA wrapper (device-only builtin) ----------------
__device__ __forceinline__ v8f wmma_bf16(v16bf a, v16bf b, v8f c) {
#if defined(__HIP_DEVICE_COMPILE__)
  return __builtin_amdgcn_wmma_f32_16x16x32_bf16(false, a, false, b, (short)0, c,
                                                 false, false);
#else
  return c;
#endif
}

// 16B global -> LDS copy; async path on gfx1250 (ASYNCcnt), sync fallback otherwise
__device__ __forceinline__ void copy16_g2l(const __bf16* gsrc, __bf16* ldst) {
#if HAVE_ASYNC_LDS
  __builtin_amdgcn_global_load_async_to_lds_b128(
      (__attribute__((address_space(1))) v4i*)gsrc,
      (__attribute__((address_space(3))) v4i*)ldst, 0, 0);
#else
  *(v8bf*)ldst = *(const v8bf*)gsrc;
#endif
}

__device__ __forceinline__ void wait_async_copies() {
#if HAVE_ASYNC_LDS
  asm volatile("s_wait_asynccnt 0x0" ::: "memory");
#endif
}

// Load a 16-element bf16 fragment as two 8-element (16B) chunks p[0..7], (p+gap)[0..7]
__device__ __forceinline__ v16bf ldfrag(const __bf16* p, int gap) {
  v8bf a = *(const v8bf*)p;
  v8bf b = *(const v8bf*)(p + gap);
  return __builtin_shufflevector(a, b, 0,1,2,3,4,5,6,7,8,9,10,11,12,13,14,15);
}

// ---------------- utility kernels ----------------
__global__ void k_zero_f32(float* p, int n) {
  int i = blockIdx.x * blockDim.x + threadIdx.x;
  if (i < n) p[i] = 0.0f;
}

__global__ void k_tobf16(const float* src, __bf16* dst, int n) {
  int i = blockIdx.x * blockDim.x + threadIdx.x;
  if (i < n) dst[i] = (__bf16)src[i];
}

// conv weight [64][64][3][3] f32  ->  WT[co][k'] bf16 with k' = (dy*3+dx)*64 + ci
__global__ void k_prep_wconv(const float* w, __bf16* wt) {
  int i = blockIdx.x * blockDim.x + threadIdx.x;
  if (i >= CH * KCONV) return;
  int co = i / KCONV, kp = i % KCONV;
  int j = kp >> 6, ci = kp & 63;
  int dy = j / 3, dx = j % 3;
  wt[i] = (__bf16)w[((co * CH + ci) * 3 + dy) * 3 + dx];
}

// MLP W1 [580][256] f32 -> WT[256][608] bf16, K permuted: k'<576: orig=c*9+j (j=k'/64,c=k'%64)
__global__ void k_prep_w1(const float* w, __bf16* wt) {
  int i = blockIdx.x * blockDim.x + threadIdx.x;
  if (i >= 256 * 608) return;
  int n = i / 608, kp = i % 608;
  float v = 0.0f;
  if (kp < 576) { int j = kp >> 6, c = kp & 63; v = w[(c * 9 + j) * 256 + n]; }
  else if (kp < 580) { v = w[kp * 256 + n]; }
  wt[i] = (__bf16)v;
}

// hidden [256][256] f32 -> WT[256][256] bf16 (transposed)
__global__ void k_prep_whid(const float* w, __bf16* wt) {
  int i = blockIdx.x * blockDim.x + threadIdx.x;
  if (i >= 256 * 256) return;
  int n = i >> 8, k = i & 255;
  wt[i] = (__bf16)w[k * 256 + n];
}

// last [256][3] f32 -> WT[16][256] bf16, rows >=3 zero
__global__ void k_prep_wlast(const float* w, __bf16* wt) {
  int i = blockIdx.x * blockDim.x + threadIdx.x;
  if (i >= 16 * 256) return;
  int n = i >> 8, k = i & 255;
  wt[i] = (__bf16)((n < 3) ? w[k * 3 + n] : 0.0f);
}

// first conv 3->64, writes padded channel-last out1 and h
__global__ void k_conv0(const float* __restrict__ x, const float* __restrict__ w,
                        const float* __restrict__ b, float* out1, float* h) {
  int i = blockIdx.x * blockDim.x + threadIdx.x;
  if (i >= NPIX * CH) return;
  int pix = i >> 6, co = i & 63;
  int y = pix / HW, xx = pix % HW;
  float acc = b[co];
  for (int ci = 0; ci < 3; ci++)
    for (int dy = 0; dy < 3; dy++)
      for (int dx = 0; dx < 3; dx++) {
        int yy = y + dy - 1, xc = xx + dx - 1;
        if (yy >= 0 && yy < HW && xc >= 0 && xc < HW)
          acc += x[ci * NPIX + yy * HW + xc] * w[((co * 3 + ci) * 3 + dy) * 3 + dx];
      }
  int off = ((y + 1) * PW + (xx + 1)) * CH + co;
  out1[off] = acc;
  h[off] = acc;
}

// ---------------- 64->64 conv as implicit GEMM (WMMA bf16) ----------------
// in/out/res: padded channel-last f32 [98][98][64]; wt: bf16 [64][576] (n-major)
// resmode: 0 none, 1: out = res + 0.1*conv, 2: out = conv + res
__global__ __launch_bounds__(256) void k_conv64(const float* __restrict__ in,
                                                float* __restrict__ out,
                                                const float* __restrict__ res,
                                                const __bf16* __restrict__ wt,
                                                const float* __restrict__ bias,
                                                int relu, int resmode) {
  __shared__ __bf16 lA[32][584];
  const int t = threadIdx.x;
  const int p0 = blockIdx.x * 32;

  // stage 32 rows x 576 K (bf16) into LDS (f32 -> bf16 conversion in flight)
  {
    int r = t >> 3, sub = t & 7;
    int pix = p0 + r;
    int y = pix / HW, xx = pix % HW;
    for (int j = 0; j < 9; j++) {
      int dy = j / 3, dx = j % 3;
      const float* src = in + ((y + dy) * PW + (xx + dx)) * CH + sub * 8;
      v4f a = *(const v4f*)src;
      v4f c = *(const v4f*)(src + 4);
      __bf16* dst = &lA[r][j * 64 + sub * 8];
      dst[0] = (__bf16)a.x; dst[1] = (__bf16)a.y; dst[2] = (__bf16)a.z; dst[3] = (__bf16)a.w;
      dst[4] = (__bf16)c.x; dst[5] = (__bf16)c.y; dst[6] = (__bf16)c.z; dst[7] = (__bf16)c.w;
    }
    if (sub == 0)
      for (int k = 576; k < 584; k++) lA[r][k] = (__bf16)0.0f;
  }
  __syncthreads();

  const int w = t >> 5, l = t & 31, lm = l & 15;
  const bool hiL = l >= 16;
  const int mt = w >> 2, nt = w & 3;

  v8f acc = {};
  for (int ks = 0; ks < 18; ks++) {
    int ka = ks * 32 + (hiL ? 8 : 0);
    v16bf af = ldfrag(&lA[mt * 16 + lm][ka], 16);
    int kb = ks * 32 + (hiL ? 8 : 0);
    const __bf16* pb = wt + (nt * 16 + lm) * KCONV + kb;
    v16bf bfr = ldfrag(pb, 16);
    acc = wmma_bf16(af, bfr, acc);
  }

  const int co = nt * 16 + lm;
  const float bv = bias[co];
#pragma unroll
  for (int v = 0; v < 8; v++) {
    int m = v + (hiL ? 8 : 0);
    int pix = p0 + mt * 16 + m;
    int y = pix / HW, xx = pix % HW;
    float val = acc[v] + bv;
    if (relu) val = fmaxf(val, 0.0f);
    int off = ((y + 1) * PW + (xx + 1)) * CH + co;
    if (resmode == 1) val = res[off] + 0.1f * val;
    else if (resmode == 2) val = val + res[off];
    out[off] = val;
  }
}

// ---------------- fused gather + 5-layer MLP (WMMA bf16) ----------------
// Each wave computes a 2x2 tile of 16x16 WMMA tiles: 2 A frags + 2 B frags -> 4 WMMAs/K-step.
__device__ __forceinline__ void mlp_layer(const __bf16* Ain, int astride, int Kdim,
                                          const __bf16* __restrict__ WT, int wstride,
                                          const float* __restrict__ bias,
                                          __bf16* Bout, int bstride, bool relu, int t) {
  const int w = t >> 5, l = t & 31, lm = l & 15;
  const bool hiL = l >= 16;
  const int ntb = w * 2;           // N tiles ntb, ntb+1 (covers 16 across 8 waves)
  v8f zero = {};
  v8f acc[4];                      // [mi*2+ni]
#pragma unroll
  for (int i = 0; i < 4; i++) acc[i] = zero;

  const int ksteps = Kdim / 32;
  for (int ks = 0; ks < ksteps; ks++) {
    int ka = ks * 32 + (hiL ? 8 : 0);
    v16bf af0 = ldfrag(Ain + lm * astride + ka, 16);
    v16bf af1 = ldfrag(Ain + (16 + lm) * astride + ka, 16);
    int kb = ks * 32 + (hiL ? 8 : 0);
    const __bf16* pb0 = WT + (ntb * 16 + lm) * wstride + kb;
    const __bf16* pb1 = WT + ((ntb + 1) * 16 + lm) * wstride + kb;
    __builtin_prefetch(pb0 + 32, 0, 1);
    v16bf bf0 = ldfrag(pb0, 16);
    v16bf bf1 = ldfrag(pb1, 16);
    acc[0] = wmma_bf16(af0, bf0, acc[0]);
    acc[1] = wmma_bf16(af0, bf1, acc[1]);
    acc[2] = wmma_bf16(af1, bf0, acc[2]);
    acc[3] = wmma_bf16(af1, bf1, acc[3]);
  }
#pragma unroll
  for (int mi = 0; mi < 2; mi++) {
#pragma unroll
    for (int ni = 0; ni < 2; ni++) {
      int n = (ntb + ni) * 16 + lm;
      float bv = bias[n];
      v8f a = acc[mi * 2 + ni];
#pragma unroll
      for (int v = 0; v < 8; v++) {
        int m = v + (hiL ? 8 : 0);
        float val = a[v] + bv;
        if (relu) val = fmaxf(val, 0.0f);
        Bout[(mi * 16 + m) * bstride + n] = (__bf16)val;
      }
    }
  }
}

__global__ __launch_bounds__(256) void k_mlp(const __bf16* __restrict__ featb,
                                             const float* __restrict__ x_coord,
                                             const float* __restrict__ x_cell,
                                             const __bf16* __restrict__ w1t, const float* __restrict__ b1,
                                             const __bf16* __restrict__ w2t, const float* __restrict__ b2,
                                             const __bf16* __restrict__ w3t, const float* __restrict__ b3,
                                             const __bf16* __restrict__ w4t, const float* __restrict__ b4,
                                             const __bf16* __restrict__ w5t, const float* __restrict__ b5,
                                             float* __restrict__ preds, float* __restrict__ areas) {
  __shared__ __bf16 sA[32][616];
  __shared__ __bf16 sB[32][264];
  __shared__ int   sIY[32], sIX[32];
  __shared__ float sR0[32], sR1[32];

  const int t = threadIdx.x;
  const int s = blockIdx.y;                  // shift 0..3
  const int qbase = blockIdx.x * 32;

  if (t < 32) {
    const float vx = (s & 2) ? 1.0f : -1.0f;
    const float vy = (s & 1) ? 1.0f : -1.0f;
    int q = qbase + t;
    float c0 = x_coord[q * 2 + 0];
    float c1 = x_coord[q * 2 + 1];
    const float rshift = 1.0f / 96.0f;
    float cy = c0 + vx * rshift + 1e-6f;
    cy = fminf(fmaxf(cy, -1.0f + 1e-6f), 1.0f - 1e-6f);
    float cx = c1 + vy * rshift + 1e-6f;
    cx = fminf(fmaxf(cx, -1.0f + 1e-6f), 1.0f - 1e-6f);
    float fy = floorf(((cy + 1.0f) * 96.0f - 1.0f) * 0.5f + 0.5f);
    float fx = floorf(((cx + 1.0f) * 96.0f - 1.0f) * 0.5f + 0.5f);
    int iy = (int)fminf(fmaxf(fy, 0.0f), 95.0f);
    int ix = (int)fminf(fmaxf(fx, 0.0f), 95.0f);
    float qcy = -1.0f + (2.0f * (float)iy + 1.0f) / 96.0f;
    float qcx = -1.0f + (2.0f * (float)ix + 1.0f) / 96.0f;
    float r0 = (c0 - qcy) * 96.0f;
    float r1 = (c1 - qcx) * 96.0f;
    sIY[t] = iy; sIX[t] = ix; sR0[t] = r0; sR1[t] = r1;
    areas[s * Q + q] = fabsf(r0 * r1) + 1e-9f;
  }
  __syncthreads();

  // gather 3x3 x 64ch bf16 features into sA via async global->LDS (16B per copy)
  {
    int r = t >> 3, sub = t & 7;
    int iy = sIY[r], ix = sIX[r];
    for (int j = 0; j < 9; j++) {
      int dy = j / 3, dx = j % 3;
      copy16_g2l(featb + ((iy + dy) * PW + (ix + dx)) * CH + sub * 8,
                 &sA[r][j * 64 + sub * 8]);
    }
    if (sub == 0) {
      int q = qbase + r;
      sA[r][576] = (__bf16)sR0[r];
      sA[r][577] = (__bf16)sR1[r];
      sA[r][578] = (__bf16)(x_cell[q * 2 + 0] * 96.0f);
      sA[r][579] = (__bf16)(x_cell[q * 2 + 1] * 96.0f);
      for (int k = 580; k < 616; k++) sA[r][k] = (__bf16)0.0f;
    }
  }
  wait_async_copies();
  __syncthreads();

  mlp_layer(&sA[0][0], 616, 608, w1t, 608, b1, &sB[0][0], 264, true, t);
  __syncthreads();
  mlp_layer(&sB[0][0], 264, 256, w2t, 256, b2, &sA[0][0], 616, true, t);
  __syncthreads();
  mlp_layer(&sA[0][0], 616, 256, w3t, 256, b3, &sB[0][0], 264, true, t);
  __syncthreads();
  mlp_layer(&sB[0][0], 264, 256, w4t, 256, b4, &sA[0][0], 616, true, t);
  __syncthreads();

  // layer 5: 256 -> 16 (3 real), waves 0..1 only
  {
    const int w = t >> 5, l = t & 31, lm = l & 15;
    const bool hiL = l >= 16;
    if (w < 2) {
      v8f acc = {};
      for (int ks = 0; ks < 8; ks++) {
        int ka = ks * 32 + (hiL ? 8 : 0);
        v16bf af = ldfrag(&sA[w * 16 + lm][ka], 16);
        int kb = ks * 32 + (hiL ? 8 : 0);
        v16bf bfr = ldfrag(w5t + lm * 256 + kb, 16);
        acc = wmma_bf16(af, bfr, acc);
      }
      if (lm < 3) {
        float bv = b5[lm];
#pragma unroll
        for (int v = 0; v < 8; v++) {
          int m = v + (hiL ? 8 : 0);
          preds[(s * Q + qbase + w * 16 + m) * 3 + lm] = acc[v] + bv;
        }
      }
    }
  }
}

// ---------------- ensemble (diagonal-swapped area weights) ----------------
__global__ void k_ensemble(const float* __restrict__ preds, const float* __restrict__ areas,
                           float* __restrict__ out) {
  int q = blockIdx.x * blockDim.x + threadIdx.x;
  if (q >= Q) return;
  float a0 = areas[q], a1 = areas[Q + q], a2 = areas[2 * Q + q], a3 = areas[3 * Q + q];
  float inv = 1.0f / (a0 + a1 + a2 + a3);
  float w0 = a3 * inv, w1 = a2 * inv, w2 = a1 * inv, w3 = a0 * inv;
#pragma unroll
  for (int ch = 0; ch < 3; ch++) {
    out[q * 3 + ch] = preds[(0 * Q + q) * 3 + ch] * w0 +
                      preds[(1 * Q + q) * 3 + ch] * w1 +
                      preds[(2 * Q + q) * 3 + ch] * w2 +
                      preds[(3 * Q + q) * 3 + ch] * w3;
  }
}

// ---------------- host ----------------
extern "C" void kernel_launch(void* const* d_in, const int* in_sizes, int n_in,
                              void* d_out, int out_size, void* d_ws, size_t ws_size,
                              hipStream_t stream) {
  (void)in_sizes; (void)n_in; (void)out_size; (void)ws_size;

  const float* x       = (const float*)d_in[0];
  const float* x_coord = (const float*)d_in[1];
  const float* x_cell  = (const float*)d_in[2];

  size_t off = 0;
  auto take = [&](size_t bytes) -> void* {
    void* p = (char*)d_ws + off;
    off += (bytes + 255) & ~(size_t)255;
    return p;
  };

  const size_t padElems = (size_t)PADPIX * CH;            // 614656
  float*  fOUT1 = (float*)take(padElems * 4);
  float*  fH    = (float*)take(padElems * 4);
  float*  fT    = (float*)take(padElems * 4);
  __bf16* featb = (__bf16*)take(padElems * 2);
  __bf16* convWT = (__bf16*)take((size_t)33 * CH * KCONV * 2);
  __bf16* w1t = (__bf16*)take((size_t)256 * 608 * 2);
  __bf16* w2t = (__bf16*)take((size_t)256 * 256 * 2);
  __bf16* w3t = (__bf16*)take((size_t)256 * 256 * 2);
  __bf16* w4t = (__bf16*)take((size_t)256 * 256 * 2);
  __bf16* w5t = (__bf16*)take((size_t)16 * 256 * 2);
  float*  preds = (float*)take((size_t)4 * Q * 3 * 4);
  float*  areas = (float*)take((size_t)4 * Q * 4);

  const int zb = (int)((padElems + 255) / 256);
  k_zero_f32<<<zb, 256, 0, stream>>>(fOUT1, (int)padElems);
  k_zero_f32<<<zb, 256, 0, stream>>>(fH, (int)padElems);
  k_zero_f32<<<zb, 256, 0, stream>>>(fT, (int)padElems);

  // weight prep
  for (int l = 0; l < 33; l++) {
    const float* w;
    if (l < 32) { int i = l >> 1; w = (const float*)d_in[5 + 4 * i + 2 * (l & 1)]; }
    else        { w = (const float*)d_in[69]; }
    k_prep_wconv<<<(CH * KCONV + 255) / 256, 256, 0, stream>>>(w, convWT + (size_t)l * CH * KCONV);
  }
  k_prep_w1<<<(256 * 608 + 255) / 256, 256, 0, stream>>>((const float*)d_in[71], w1t);
  k_prep_whid<<<256, 256, 0, stream>>>((const float*)d_in[73], w2t);
  k_prep_whid<<<256, 256, 0, stream>>>((const float*)d_in[75], w3t);
  k_prep_whid<<<256, 256, 0, stream>>>((const float*)d_in[77], w4t);
  k_prep_wlast<<<16, 256, 0, stream>>>((const float*)d_in[79], w5t);

  // encoder
  k_conv0<<<(NPIX * CH + 255) / 256, 256, 0, stream>>>(x, (const float*)d_in[3],
                                                       (const float*)d_in[4], fOUT1, fH);
  const int convGrid = NPIX / 32;  // 288
  for (int i = 0; i < 16; i++) {
    const float* bA = (const float*)d_in[6 + 4 * i];
    const float* bB = (const float*)d_in[8 + 4 * i];
    k_conv64<<<convGrid, 256, 0, stream>>>(fH, fT, fH, convWT + (size_t)(2 * i) * CH * KCONV, bA, 1, 0);
    k_conv64<<<convGrid, 256, 0, stream>>>(fT, fH, fH, convWT + (size_t)(2 * i + 1) * CH * KCONV, bB, 0, 1);
  }
  k_conv64<<<convGrid, 256, 0, stream>>>(fH, fT, fOUT1, convWT + (size_t)32 * CH * KCONV,
                                         (const float*)d_in[70], 0, 2);
  k_tobf16<<<zb, 256, 0, stream>>>(fT, featb, (int)padElems);

  // fused gather + MLP, 4 shifts
  dim3 mlpGrid(Q / 32, 4);
  k_mlp<<<mlpGrid, 256, 0, stream>>>(featb, x_coord, x_cell,
                                     w1t, (const float*)d_in[72],
                                     w2t, (const float*)d_in[74],
                                     w3t, (const float*)d_in[76],
                                     w4t, (const float*)d_in[78],
                                     w5t, (const float*)d_in[80],
                                     preds, areas);

  k_ensemble<<<Q / 256, 256, 0, stream>>>(preds, areas, (float*)d_out);
}
